// Rewirescorelayer_61297773248645
// MI455X (gfx1250) — compile-verified, matrix operand order checked by reading, not run on hardware
//
#include <hip/hip_runtime.h>

typedef __attribute__((ext_vector_type(16))) _Float16 v16h;
typedef __attribute__((ext_vector_type(8)))  float    v8f;
typedef __attribute__((ext_vector_type(4)))  float    f4;

#define Nn   4096
#define INF_ 512
#define HD   1024      // Q|K concat width
#define WW   128
#define SPAN 160       // column span covered per 16-node block (<=144 needed)
#define INV_TAU 2.0f

// ---------------- prep: fp32 -> f16 convert, weight transpose, bias concat ---------
__global__ void prep_kernel(const float* __restrict__ X,
                            const float* __restrict__ Wq, const float* __restrict__ bq,
                            const float* __restrict__ Wk, const float* __restrict__ bk,
                            _Float16* __restrict__ Xh, _Float16* __restrict__ WT,
                            float* __restrict__ biasc) {
    int idx = blockIdx.x * 256 + threadIdx.x;
    if (idx < Nn * INF_) {                       // X convert: [4096][512]
        Xh[idx] = (_Float16)X[idx];
        return;
    }
    idx -= Nn * INF_;
    if (idx < 1024 * 512) {                      // WT[n][k] = W[k][n]; n<512 -> Wq, else Wk
        int n = idx >> 9, k = idx & 511;
        float v = (n < 512) ? Wq[k * 512 + n] : Wk[k * 512 + (n - 512)];
        WT[idx] = (_Float16)v;
        return;
    }
    idx -= 1024 * 512;
    if (idx < 1024)
        biasc[idx] = (idx < 512) ? bq[idx] : bk[idx - 512];
}

// -------- GEMM: QKh[4096][1024](f16) = Xh @ [Wq|Wk] + bias, WMMA f16, f32 accum ----
__global__ __launch_bounds__(256) void gemm_kernel(const _Float16* __restrict__ Xh,
                                                   const _Float16* __restrict__ WT,
                                                   const float* __restrict__ biasc,
                                                   _Float16* __restrict__ QKh) {
    const int wave  = (blockIdx.x * 256 + threadIdx.x) >> 5;   // 16384 waves
    const int lane  = threadIdx.x & 31;
    const int tileM = wave >> 6;     // 0..255
    const int tileN = wave & 63;     // 0..63
    const int rsel  = lane & 15;
    const int kOff  = (lane >> 4) << 4;   // 0 or 16

    const _Float16* aBase = Xh + (size_t)(tileM * 16 + rsel) * INF_ + kOff;
    const _Float16* bBase = WT + (size_t)(tileN * 16 + rsel) * INF_ + kOff;

    v8f acc = {};
#pragma unroll
    for (int k0 = 0; k0 < INF_; k0 += 32) {
        v16h a = *(const v16h*)(aBase + k0);
        v16h b = *(const v16h*)(bBase + k0);
        acc = __builtin_amdgcn_wmma_f32_16x16x32_f16(false, a, false, b,
                                                     (short)0, acc, false, false);
    }
    const int col     = tileN * 16 + rsel;
    const int rowBase = tileM * 16 + ((lane >> 4) << 3);
    const float bias  = biasc[col];
#pragma unroll
    for (int r = 0; r < 8; r++)
        QKh[(size_t)(rowBase + r) * HD + col] = (_Float16)(acc[r] + bias);
}

// -------- windowed attention via WMMA: attn[N][128] --------------------------------
// Block = 16 consecutive nodes. Scores for the shared column span [cbase, cbase+160)
// computed on matrix pipes; per-head masked softmax + head mean in LDS.
__global__ __launch_bounds__(256) void attn_wmma_kernel(const _Float16* __restrict__ QKh,
                                                        const int* __restrict__ bounds,
                                                        float* __restrict__ attn) {
    const int n0   = blockIdx.x * 16;
    const int tid  = threadIdx.x;
    const int lane = tid & 31;
    const int wv   = tid >> 5;            // 8 waves
    const int rsel = lane & 15;
    const int kOff = (lane >> 4) << 4;

    __shared__ float sc[4 * 16 * SPAN];   // 4 heads' scores: [hh][node][col]  (40 KB)
    __shared__ float attnAcc[16 * WW];    // head-prob accumulator            ( 8 KB)
    __shared__ float mh[64], sh[64];      // per (node, head-in-phase) stats

    const int b0 = bounds[0], b1 = bounds[1], b2 = bounds[2], b3 = bounds[3];
    const int posB  = (n0 >= b0) + (n0 >= b1) + (n0 >= b2) + (n0 >= b3);
    const int segsB = (posB == 0) ? 0 : bounds[posB - 1];
    const int cbase = max(segsB, n0 - 64);

    for (int i = tid; i < 16 * WW; i += 256) attnAcc[i] = 0.f;

    for (int hp = 0; hp < 2; hp++) {
        const int h  = hp * 4 + (wv & 3); // global head
        const int hh = wv & 3;            // head slot in this phase
        const int t0 = (wv >> 2) * 5;     // 2 waves split the 10 column tiles
        __syncthreads();                  // sc safe to overwrite

        // A fragments: Q rows n0..n0+15, dims h*64..h*64+63 (reused for all tiles)
        const _Float16* aPtr = QKh + (size_t)(n0 + rsel) * HD + h * 64 + kOff;
        v16h a0 = *(const v16h*)aPtr;
        v16h a1 = *(const v16h*)(aPtr + 32);

        for (int tt = 0; tt < 5; tt++) {
            const int t = t0 + tt;
            int colRow = cbase + t * 16 + rsel;
            colRow = (colRow < Nn) ? colRow : (Nn - 1);   // clamp; masked later
            const _Float16* bPtr = QKh + (size_t)colRow * HD + 512 + h * 64 + kOff;
            v16h bf0 = *(const v16h*)bPtr;
            v16h bf1 = *(const v16h*)(bPtr + 32);
            v8f acc = {};
            acc = __builtin_amdgcn_wmma_f32_16x16x32_f16(false, a0, false, bf0,
                                                         (short)0, acc, false, false);
            acc = __builtin_amdgcn_wmma_f32_16x16x32_f16(false, a1, false, bf1,
                                                         (short)0, acc, false, false);
            const int col   = t * 16 + rsel;
            const int rBase = (lane >> 4) << 3;
#pragma unroll
            for (int r = 0; r < 8; r++)
                sc[(hh * 16 + rBase + r) * SPAN + col] = acc[r];
        }
        __syncthreads();

        // per-(node, head) masked softmax stats; logits = score * (1/sqrt(64)/TAU)
        if (tid < 64) {
            const int r = tid >> 2, h2 = tid & 3;
            const int n = n0 + r;
            const int pos   = (n >= b0) + (n >= b1) + (n >= b2) + (n >= b3);
            const int segs  = (pos == 0) ? 0 : bounds[pos - 1];
            const int sege  = bounds[(pos < 3) ? pos : 3];
            const int start = max(segs, n - 64);
            const int wsize = min(sege, n + 64) - start;
            const int o     = start - cbase;
            const float* row = &sc[(h2 * 16 + r) * SPAN];
            float m = -INFINITY;
            for (int c = 0; c < wsize; c++) m = fmaxf(m, row[o + c] * 0.25f);
            float su = 0.f;
            for (int c = 0; c < wsize; c++) su += __expf(row[o + c] * 0.25f - m);
            mh[r * 4 + h2] = m; sh[r * 4 + h2] = su;
        }
        __syncthreads();

        // accumulate probabilities for this phase's 4 heads (thread owns its (r,w) set)
#pragma unroll
        for (int i = 0; i < 8; i++) {
            const int e = tid + i * 256;
            const int r = e >> 7, w = e & 127;
            const int n = n0 + r;
            const int pos   = (n >= b0) + (n >= b1) + (n >= b2) + (n >= b3);
            const int segs  = (pos == 0) ? 0 : bounds[pos - 1];
            const int sege  = bounds[(pos < 3) ? pos : 3];
            const int start = max(segs, n - 64);
            const int wsize = min(sege, n + 64) - start;
            if (w < wsize) {
                const int o = start - cbase;
                float a = 0.f;
#pragma unroll
                for (int h2 = 0; h2 < 4; h2++)
                    a += __expf(sc[(h2 * 16 + r) * SPAN + (w + o)] * 0.25f - mh[r * 4 + h2])
                         / sh[r * 4 + h2];
                attnAcc[e] += a;
            }
        }
    }
    __syncthreads();
    for (int i = tid; i < 16 * WW; i += 256)
        attn[(size_t)n0 * WW + i] = attnAcc[i] * 0.125f;   // mean over 8 heads
}

// ------- fused scatter + gumbel softmax over dense rows (HBM-bound, NT hints) ------
__global__ __launch_bounds__(256) void gumbel_kernel(const float* __restrict__ g,
                                                     const float* __restrict__ attn,
                                                     const int* __restrict__ bounds,
                                                     float* __restrict__ out) {
    const int n = blockIdx.x;
    const int t = threadIdx.x;            // holds 16 columns in registers
    __shared__ float aw[WW];
    __shared__ float red[256];

    if (t < WW) aw[t] = attn[(size_t)n * WW + t];
    const int b0 = bounds[0], b1 = bounds[1], b2 = bounds[2], b3 = bounds[3];
    const int pos   = (n >= b0) + (n >= b1) + (n >= b2) + (n >= b3);
    const int segs  = (pos == 0) ? 0 : bounds[pos - 1];
    const int start = max(segs, n - 64);
    __syncthreads();

    const f4* grow = (const f4*)(g + (size_t)n * Nn);
    f4 v[4];
    float lmax = -INFINITY;
#pragma unroll
    for (int c = 0; c < 4; c++) {
        f4 gv = __builtin_nontemporal_load(&grow[c * 256 + t]);   // streaming read
        const int j = c * 1024 + t * 4;
        float* vp = (float*)&v[c];
        float gg[4] = {gv.x, gv.y, gv.z, gv.w};
#pragma unroll
        for (int e = 0; e < 4; e++) {
            int w = j + e - start;
            float dense = (w >= 0 && w < WW) ? aw[w] : 0.f;
            float val = (gg[e] + dense) * INV_TAU;
            vp[e] = val;
            lmax = fmaxf(lmax, val);
        }
    }
    red[t] = lmax; __syncthreads();
    for (int s2 = 128; s2 > 0; s2 >>= 1) {
        if (t < s2) red[t] = fmaxf(red[t], red[t + s2]);
        __syncthreads();
    }
    const float m = red[0]; __syncthreads();

    float lsum = 0.f;
#pragma unroll
    for (int c = 0; c < 4; c++) {
        float* vp = (float*)&v[c];
#pragma unroll
        for (int e = 0; e < 4; e++) { float ex = __expf(vp[e] - m); vp[e] = ex; lsum += ex; }
    }
    red[t] = lsum; __syncthreads();
    for (int s2 = 128; s2 > 0; s2 >>= 1) {
        if (t < s2) red[t] += red[t + s2];
        __syncthreads();
    }
    const float inv = 1.0f / red[0];

    f4* orow = (f4*)(out + (size_t)n * Nn);
#pragma unroll
    for (int c = 0; c < 4; c++) {
        float* vp = (float*)&v[c];
        f4 ov = {vp[0] * inv, vp[1] * inv, vp[2] * inv, vp[3] * inv};
        __builtin_nontemporal_store(ov, &orow[c * 256 + t]);      // streaming write
    }
}

// ---------------- launch -----------------------------------------------------------
extern "C" void kernel_launch(void* const* d_in, const int* in_sizes, int n_in,
                              void* d_out, int out_size, void* d_ws, size_t ws_size,
                              hipStream_t stream) {
    const float* X      = (const float*)d_in[0];
    const float* Wq     = (const float*)d_in[1];
    const float* bq     = (const float*)d_in[2];
    const float* Wk     = (const float*)d_in[3];
    const float* bk     = (const float*)d_in[4];
    const float* g      = (const float*)d_in[5];
    const int*   bounds = (const int*)d_in[6];

    char* ws = (char*)d_ws;
    _Float16* Xh    = (_Float16*)(ws);                 // 4096*512*2   = 4,194,304 B
    _Float16* WT    = (_Float16*)(ws + 4194304);       // 1024*512*2   = 1,048,576 B
    float*    biasc = (float*)   (ws + 5242880);       // 1024*4       =     4,096 B
    _Float16* QKh   = (_Float16*)(ws + 5246976);       // 4096*1024*2  = 8,388,608 B
    float*    attn  = (float*)   (ws + 13635584);      // 4096*128*4   = 2,097,152 B

    prep_kernel     <<<10244, 256, 0, stream>>>(X, Wq, bq, Wk, bk, Xh, WT, biasc);
    gemm_kernel     <<<2048,  256, 0, stream>>>(Xh, WT, biasc, QKh);
    attn_wmma_kernel<<<256,   256, 0, stream>>>(QKh, bounds, attn);
    gumbel_kernel   <<<4096,  256, 0, stream>>>(g, attn, bounds, (float*)d_out);
}
